// GroupedESNCell_24257975288439
// MI455X (gfx1250) — matrix operand order repaired and checked by reading, not compiled, not run on hardware
//
#include <hip/hip_runtime.h>
#include <hip/hip_bf16.h>

typedef __attribute__((ext_vector_type(16))) __bf16       v16bf;
typedef __attribute__((ext_vector_type(8)))  float        v8f;
typedef __attribute__((ext_vector_type(8)))  unsigned int v8u;

#define G_ 8
#define H_ 2048
#define T_ 4096
#define WGS_PER_GROUP 16
#define NWG (G_ * WGS_PER_GROUP)        // 128 workgroups (persistent, global barrier)
#define THREADS 256                      // 8 waves (wave32)
#define KT_ITERS (H_ / 32)               // 64 WMMA k-steps per mat-vec
#define ROWS_PER_WG (H_ / WGS_PER_GROUP) // 128

// d_ws layout: [0..1023] barrier counter | [1KB..129KB) pre double-buffer | [256KB..) bf16 weights
#define WS_PRE_OFF 1024
#define WS_WT_OFF  (256 * 1024)          // needs ws_size >= 256KB + 64MB

// ---------------------------------------------------------------------------
// One-time: repack W_hh (fp32, [G,H,H]) into bf16 WMMA-A-fragment tiles.
// Tile layout: [g][rowTile(128)][kt(64)][lane(32)][e(16)] so the scan kernel
// loads 32 contiguous bytes per lane == exact A-fragment register layout:
//   lane<16 : M = rt*16 + lane,    K = kt*32 + (e<8 ? e : e+8)
//   lane>=16: M = rt*16 + lane-16, K = kt*32 + (e<8 ? e : e+8) + 8
// (ISA 7.12.2, 16-bit A-matrix 16x32 table.)
// ---------------------------------------------------------------------------
__global__ __launch_bounds__(256) void esn_pack_w(const float* __restrict__ Whh,
                                                  __bf16* __restrict__ Wt) {
  size_t o = (size_t)blockIdx.x * 256 + threadIdx.x;
  int e    = (int)(o & 15);
  int lane = (int)((o >> 4) & 31);
  int kt   = (int)((o >> 9) & 63);
  int rtg  = (int)(o >> 15);             // g*128 + rowTile
  int g    = rtg >> 7;
  int rt   = rtg & 127;
  int row  = rt * 16 + (lane & 15);
  int K    = kt * 32 + ((e < 8) ? e : e + 8) + (lane >> 4) * 8;
  Wt[o] = (__bf16)Whh[((size_t)g * H_ + row) * H_ + K];
}

__global__ void esn_init(unsigned int* ctr) {
  if (threadIdx.x == 0) *ctr = 0u;       // ws is poisoned 0xAA; re-zero every call
}

// ---------------------------------------------------------------------------
// Persistent scan kernel: block b -> (group g = b/16, slice = b%16).
// Groups are independent recurrences; WGs within a group publish their pre
// slice and sync per step via a monotonic atomic-counter barrier so every WG
// can form the (deterministic) group norm.
// ---------------------------------------------------------------------------
__global__ __launch_bounds__(THREADS) void esn_scan(const float* __restrict__ x,
                                                    const float* __restrict__ Wih,
                                                    const __bf16* __restrict__ Wt,
                                                    float* __restrict__ out,
                                                    float* __restrict__ preBuf,
                                                    unsigned int* __restrict__ gctr) {
  __shared__ float        h_s[H_];
  __shared__ unsigned int hpk_s[H_];     // (bf16 | bf16<<16) packed pairs of h
  __shared__ float        wih_s[H_];
  __shared__ float        red_s[THREADS / 32];

  const int tid   = threadIdx.x;
  const int lane  = tid & 31;
  const int wave  = tid >> 5;
  const int g     = blockIdx.x / WGS_PER_GROUP;
  const int slice = blockIdx.x % WGS_PER_GROUP;
  const int rt    = slice * 8 + wave;    // this wave's 16-row tile within the group
  const int rowBase = rt * 16;

  for (int r = tid; r < H_; r += THREADS) {
    h_s[r]   = 0.0f;
    hpk_s[r] = 0u;
    wih_s[r] = Wih[r];
  }
  __syncthreads();

  const float lr  = 1.0f * (float)(G_ - g) / (float)G_;  // LEAKY=1.0
  const float oml = 1.0f - lr;
  const __bf16* Ap = Wt + (size_t)(g * 128 + rt) * (KT_ITERS * 512) + lane * 16;

  for (int t = 0; t < T_; ++t) {
    // ---- pre = W_hh[g] @ h : 64 chained bf16 WMMAs (fp32 accumulate) ----
    v8f acc = {};
    const __bf16* ap = Ap;
#pragma unroll 4
    for (int kt = 0; kt < KT_ITERS; ++kt) {
      v16bf a = *(const v16bf*)ap;                       // 2x global_load_b128, coalesced
      ap += 512;
      // B (32x16, elems index K in A's table order, lanes index N): h replicated
      // across all 16 N-columns, so every column of D equals pre.
      v8u bu;
#pragma unroll
      for (int e2 = 0; e2 < 8; ++e2) {
        int k0 = kt * 32 + (lane >> 4) * 8 + 2 * e2 + (e2 >= 4 ? 8 : 0);
        bu[e2] = (hpk_s[k0] & 0xFFFFu) | (hpk_s[k0 + 1] & 0xFFFF0000u);
      }
      v16bf b = __builtin_bit_cast(v16bf, bu);
      acc = __builtin_amdgcn_wmma_f32_16x16x32_bf16(false, a, false, b,
                                                    (short)0, acc, false, false);
    }

    // ---- publish this wave's 16 rows of pre (+ input drive) ----
    float* preG = preBuf + (size_t)(t & 1) * (G_ * H_) + (size_t)g * H_;
    if ((lane & 15) == 0) {                              // lanes 0 / 16 hold rows v + 8*(lane>>4)
      float xt = x[t];
      int rb = rowBase + (lane >> 4) * 8;
#pragma unroll
      for (int v = 0; v < 8; ++v) preG[rb + v] = acc[v] + xt * wih_s[rb + v];
    }
    __threadfence();                                     // release pre to device scope
    __syncthreads();

    // ---- global step barrier: monotonic counter, no resets ----
    if (tid == 0) {
      __hip_atomic_fetch_add(gctr, 1u, __ATOMIC_ACQ_REL, __HIP_MEMORY_SCOPE_AGENT);
      const unsigned int target = (unsigned int)NWG * (unsigned int)(t + 1);
      while (__hip_atomic_load(gctr, __ATOMIC_ACQUIRE, __HIP_MEMORY_SCOPE_AGENT) < target)
        __builtin_amdgcn_s_sleep(2);
    }
    __syncthreads();
    __threadfence();                                     // acquire: invalidate stale WGP$

    // ---- read full pre (vectorized), deterministic norm ----
    const int r0 = tid * 8;
    float4 p4a = *(const float4*)(preG + r0);            // 2x global_load_b128
    float4 p4b = *(const float4*)(preG + r0 + 4);
    float pv[8] = {p4a.x, p4a.y, p4a.z, p4a.w, p4b.x, p4b.y, p4b.z, p4b.w};
    float ss = 0.0f;
#pragma unroll
    for (int j = 0; j < 8; ++j) ss += pv[j] * pv[j];
    // wave32 butterfly reduce (fixed order -> bit-deterministic, same in all WGs)
#pragma unroll
    for (int m = 16; m > 0; m >>= 1) ss += __shfl_xor(ss, m, 32);
    if (lane == 0) red_s[wave] = ss;
    __syncthreads();
    float sum = 0.0f;
#pragma unroll
    for (int w = 0; w < THREADS / 32; ++w) sum += red_s[w]; // fixed order
    const float nrm = sqrtf(sum);
    const float fac = lr * 100.0f / nrm;                 // RADIUS = 100

    // ---- h_new = (1-lr)*h + fac*pre ; refresh fp32 + packed-bf16 copies ----
    float hv[8];
#pragma unroll
    for (int j = 0; j < 8; ++j) {
      int r = r0 + j;
      float hn = oml * h_s[r] + fac * pv[j];
      hv[j] = hn;
      h_s[r] = hn;
      unsigned short hb = __builtin_bit_cast(unsigned short, (__bf16)hn);
      hpk_s[r] = (unsigned int)hb * 0x10001u;            // low|high duplicated
    }

    // ---- this WG writes only its 128-row output slice ----
    const int sBase = slice * ROWS_PER_WG;
    if (r0 >= sBase && r0 < sBase + ROWS_PER_WG) {
      float* op = out + (size_t)t * (G_ * H_) + (size_t)g * H_ + r0;
#pragma unroll
      for (int j = 0; j < 8; ++j) op[j] = hv[j];
    }
    __syncthreads();                                     // hpk_s stable before next WMMA sweep
  }
}

extern "C" void kernel_launch(void* const* d_in, const int* in_sizes, int n_in,
                              void* d_out, int out_size, void* d_ws, size_t ws_size,
                              hipStream_t stream) {
  (void)in_sizes; (void)n_in; (void)out_size; (void)ws_size;
  const float* x   = (const float*)d_in[0];   // [T,1]
  const float* Wih = (const float*)d_in[1];   // [H,1]
  const float* Whh = (const float*)d_in[2];   // [G,H,H]
  float* out = (float*)d_out;                 // [T, G*H] fp32

  uint8_t* ws = (uint8_t*)d_ws;
  unsigned int* gctr = (unsigned int*)ws;
  float*  preBuf = (float*)(ws + WS_PRE_OFF);
  __bf16* Wt     = (__bf16*)(ws + WS_WT_OFF); // 64 MB bf16 weights (L2-resident)

  esn_init<<<1, 64, 0, stream>>>(gctr);
  const size_t total = (size_t)G_ * H_ * H_;  // 33.5M elements
  esn_pack_w<<<(unsigned)(total / 256), 256, 0, stream>>>(Whh, Wt);
  esn_scan<<<NWG, THREADS, 0, stream>>>(x, Wih, Wt, out, preBuf, gctr);
}
